// RoPEAttention_46377056862242
// MI455X (gfx1250) — matrix operand order repaired and checked by reading, not compiled, not run on hardware
//
#include <hip/hip_runtime.h>
#include <hip/hip_bf16.h>

// ---------------------------------------------------------------------------
// RoPE attention for MI455X (gfx1250, wave32, WMMA).
// ~38 GFLOP vs ~40MB traffic @ 23.3 TB/s -> matrix-op bound, so everything
// routes through v_wmma_f32_16x16x32_bf16 (bf16 operands, f32 accum).
// Weights pre-converted to bf16 once; K/V staged per-block into LDS via the
// gfx1250 async-to-LDS path (ASYNCcnt) when available.
// ---------------------------------------------------------------------------

#define Dm   256
#define Hh   8
#define HDm  32
#define NIMGm 4096
#define Nm   4112          // NIMG + NTC, 257 tiles of 16
#define Bm   2
#define NKEm 64
#define TOKS (Bm*Nm)       // 8224 flat rows
#define MT   (TOKS/16)     // 514 row tiles
#define QT_PER_BH (Nm/16)  // 257 query tiles per (b,h)
#define SCALE_Q 0.17677669529663687f  // 1/sqrt(32)

// workspace byte offsets (256B-aligned); total ~20.6 MiB
#define OFF_K  ((size_t)4210688)    // Q/K: B*H*N*HD bf16 = 4,210,688 B each
#define OFF_V  ((size_t)8421376)    // V transposed [b,h,hd,n] bf16
#define OFF_O  ((size_t)12632064)   // B*N*D f32 = 8,421,376 B
#define OFF_W  ((size_t)21053440)   // 4 x 256x256 bf16 = 131,072 B each

typedef __attribute__((ext_vector_type(16))) __bf16 v16bf;
typedef __attribute__((ext_vector_type(8)))  float  v8f;

union V16 {
  v16bf v;
  uint4 q[2];
  unsigned short u[16];
};

static __device__ __forceinline__ unsigned short f2bf_bits(float f) {
  union { __bf16 b; unsigned short u; } cv;
  cv.b = (__bf16)f;                      // native cvt (RNE)
  return cv.u;
}

static __device__ __forceinline__ void pack8(V16& d, int base, float4 a, float4 b) {
  d.u[base+0] = f2bf_bits(a.x); d.u[base+1] = f2bf_bits(a.y);
  d.u[base+2] = f2bf_bits(a.z); d.u[base+3] = f2bf_bits(a.w);
  d.u[base+4] = f2bf_bits(b.x); d.u[base+5] = f2bf_bits(b.y);
  d.u[base+6] = f2bf_bits(b.z); d.u[base+7] = f2bf_bits(b.w);
}

static __device__ __forceinline__ v8f bfwmma(v16bf a, v16bf b, v8f c) {
  return __builtin_amdgcn_wmma_f32_16x16x32_bf16(false, a, false, b,
                                                 (short)0, c, false, false);
}

static __device__ __forceinline__ float rmax16(float v) {
  v = fmaxf(v, __shfl_xor(v, 1, 32));
  v = fmaxf(v, __shfl_xor(v, 2, 32));
  v = fmaxf(v, __shfl_xor(v, 4, 32));
  v = fmaxf(v, __shfl_xor(v, 8, 32));
  return v;
}
static __device__ __forceinline__ float rsum16(float v) {
  v += __shfl_xor(v, 1, 32);
  v += __shfl_xor(v, 2, 32);
  v += __shfl_xor(v, 4, 32);
  v += __shfl_xor(v, 8, 32);
  return v;
}

// gfx1250 async global->LDS staging (ASYNCcnt), feature-guarded.
// Builtin parameter type (from clang diagnostic): raw GNU vector
// 'int __attribute__((vector_size(16)))' pointers in AS1/AS3.
#if __has_builtin(__builtin_amdgcn_global_load_async_to_lds_b128)
#define HAVE_ASYNC 1
typedef int v4i_raw __attribute__((vector_size(16)));
typedef __attribute__((address_space(1))) v4i_raw gas_v4i;
typedef __attribute__((address_space(3))) v4i_raw las_v4i;
#define STAGE16(gsrc, ldst)                                                   \
  __builtin_amdgcn_global_load_async_to_lds_b128(                             \
      (gas_v4i*)(void*)(gsrc), (las_v4i*)(void*)(ldst), 0, 0)
static __device__ __forceinline__ void wait_async0() {
#if __has_builtin(__builtin_amdgcn_s_wait_asynccnt)
  __builtin_amdgcn_s_wait_asynccnt(0);
#else
  asm volatile("s_wait_asynccnt 0x0" ::: "memory");
#endif
}
#else
#define HAVE_ASYNC 0
#define STAGE16(gsrc, ldst) (*(uint4*)(ldst) = *(const uint4*)(gsrc))
static __device__ __forceinline__ void wait_async0() {}
#endif

// ---------------------------------------------------------------------------
// wcvt_kernel: one-time f32 -> bf16 conversion of Wq/Wk/Wv/Wo into workspace.
// ---------------------------------------------------------------------------
__global__ __launch_bounds__(256) void wcvt_kernel(
    const float* w0, const float* w1, const float* w2, const float* w3,
    char* ws) {
  const float* src = (blockIdx.y == 0) ? w0 :
                     (blockIdx.y == 1) ? w1 :
                     (blockIdx.y == 2) ? w2 : w3;
  unsigned short* dst =
      (unsigned short*)(ws + OFF_W) + (size_t)blockIdx.y * (Dm * Dm);
  int i = (blockIdx.x * 256 + threadIdx.x) * 8;    // grid.x=32 -> 65536 elems
  const float4* sp = (const float4*)(src + i);
  float4 a = sp[0], b = sp[1];
  V16 t;
  pack8(t, 0, a, b);
  *(uint4*)(dst + i) = t.q[0];
}

// ---------------------------------------------------------------------------
// proj_kernel: (16-row tile, 256 cols) = in @ W^T + bias, then:
//   mode 0: RoPE * 1/sqrt(HD) -> bf16 Q[b,h,n,hd]
//   mode 1: RoPE              -> bf16 K[b,h,n,hd]
//   mode 2:                   -> bf16 V transposed [b,h,hd,n]
//   mode 3:                   -> f32 `out` (final O @ Wo^T + bo)
// W is pre-converted bf16 (row-major 256x256). 8 waves; wave w owns cols
// [32w, 32w+32).
// ---------------------------------------------------------------------------
__global__ __launch_bounds__(256) void proj_kernel(
    const float* in, const unsigned short* W, const float* bias,
    const float* fimg, const float* ftxt,
    char* ws, float* out, int mode) {

  __shared__ __align__(16) unsigned short sA[16 * 264];  // 16x256 bf16 + pad

  const int tid   = threadIdx.x;
  const int lane  = tid & 31;
  const int wave  = tid >> 5;
  const int lj    = lane & 15;
  const int lhalf = lane >> 4;
  const int tile  = blockIdx.x;

  // stage A tile (16x256 f32 -> bf16 LDS), 16 elements per thread
  {
    int row = tid >> 4;
    int cb  = (tid & 15) * 16;
    const float4* src = (const float4*)(in + (size_t)(tile * 16 + row) * Dm + cb);
    float4 a0 = src[0], a1 = src[1], a2 = src[2], a3 = src[3];
    V16 t;
    pack8(t, 0, a0, a1);
    pack8(t, 8, a2, a3);
    *(uint4*)(sA + row * 264 + cb)     = t.q[0];
    *(uint4*)(sA + row * 264 + cb + 8) = t.q[1];
  }
  __syncthreads();

  const int kbA  = lhalf * 8;    // A-operand K base within 32-K step
  const int kbB  = lhalf * 8;    // B-operand K base
  const int bidx = tile / QT_PER_BH;
  const int nt   = tile % QT_PER_BH;

  for (int ct = 0; ct < 2; ++ct) {
    const int dg = wave * 32 + ct * 16 + lj;   // this lane's output column

    v8f c;
    {
      float bv = bias[dg];
      #pragma unroll
      for (int r = 0; r < 8; ++r) c[r] = bv;
    }

    #pragma unroll
    for (int ks = 0; ks < 8; ++ks) {
      V16 a;
      a.q[0] = *(const uint4*)(sA + lj * 264 + ks * 32 + kbA);
      a.q[1] = *(const uint4*)(sA + lj * 264 + ks * 32 + kbA + 16);
      V16 bm;
      const unsigned short* wp = W + (size_t)dg * Dm + ks * 32 + kbB;
      bm.q[0] = *(const uint4*)wp;
      bm.q[1] = *(const uint4*)(wp + 16);
      c = bfwmma(a.v, bm.v, c);
    }

    const float qscale = (mode == 0) ? SCALE_Q : 1.0f;
    #pragma unroll
    for (int r = 0; r < 8; ++r) {
      const int row = r + lhalf * 8;
      const int n   = nt * 16 + row;
      float val = c[r];
      if (mode <= 1) {
        // RoPE: channel pairs (2i,2i+1) are adjacent lanes in C layout
        const int p = (dg & (HDm - 1)) >> 1;
        const float* fr = (n < NIMGm)
            ? (fimg + ((size_t)n * (HDm / 2) + p) * 2)
            : (ftxt + ((size_t)(n - NIMGm) * (HDm / 2) + p) * 2);
        float cs = fr[0], sn = fr[1];
        float other = __shfl_xor(val, 1, 32);
        val = (dg & 1) ? (other * sn + val * cs) : (val * cs - other * sn);
        val *= qscale;
      }
      const int hh = dg >> 5, hd = dg & 31;
      if (mode == 0 || mode == 1) {
        unsigned short* dst =
            (unsigned short*)(ws + (mode == 0 ? (size_t)0 : OFF_K));
        dst[((size_t)(bidx * Hh + hh) * Nm + n) * HDm + hd] = f2bf_bits(val);
      } else if (mode == 2) {
        unsigned short* dst = (unsigned short*)(ws + OFF_V);
        dst[((size_t)(bidx * Hh + hh) * HDm + hd) * Nm + n] = f2bf_bits(val);
      } else {
        out[(size_t)(tile * 16 + row) * Dm + dg] = val;
      }
    }
  }
}

// ---------------------------------------------------------------------------
// attn_kernel: per (b,h), flash attention; 8 waves/block, one 16-query tile
// per wave. K/V key-blocks (32 keys) are staged cooperatively into LDS once
// per block (async-to-LDS when available) and shared by all 8 waves.
// Pass 0: keys [0, N-NKE); pass 1: keys [N-NKE, N); separate softmaxes, summed.
// ---------------------------------------------------------------------------
__global__ __launch_bounds__(256) void attn_kernel(char* ws) {
  // padded rows (40 ushorts = 80B) -> conflict-free 16-lane b128 reads
  __shared__ __align__(16) unsigned short sK[32 * 40];
  __shared__ __align__(16) unsigned short sV[32 * 40];
  __shared__ __align__(16) unsigned short sPu[8 * 16 * 40];

  const int tid   = threadIdx.x;
  const int lane  = tid & 31;
  const int wave  = tid >> 5;
  const int lj    = lane & 15;
  const int lhalf = lane >> 4;

  const int qt     = blockIdx.x * 8 + wave;
  const bool active = (qt < QT_PER_BH);          // wave-uniform
  const int b = blockIdx.y >> 3, h = blockIdx.y & 7;
  const size_t bh = (size_t)(b * Hh + h);

  const unsigned short* Qb = (const unsigned short*)ws;
  const unsigned short* Kb = (const unsigned short*)(ws + OFF_K);
  const unsigned short* Vt = (const unsigned short*)(ws + OFF_V);
  float* O = (float*)(ws + OFF_O);
  unsigned short* sPw = sPu + wave * 640;

  // Q tile in A-operand layout (1/sqrt(HD) folded in at projection)
  V16 qa;
  if (active) {
    const unsigned short* qp =
        Qb + (bh * Nm + (size_t)(qt * 16 + lj)) * HDm + lhalf * 8;
    qa.q[0] = *(const uint4*)qp;
    qa.q[1] = *(const uint4*)(qp + 16);
  }

  float out1[8], out2[8];
  #pragma unroll
  for (int r = 0; r < 8; ++r) { out1[r] = 0.f; out2[r] = 0.f; }

  const int kbB = lhalf * 8;
  const int kbA = lhalf * 8;

  for (int pass = 0; pass < 2; ++pass) {
    const int k0 = pass ? (Nm - NKEm) : 0;
    const int k1 = pass ? Nm : (Nm - NKEm);

    float m[8], l[8];
    v8f acc1 = {}, acc2 = {};
    #pragma unroll
    for (int r = 0; r < 8; ++r) { m[r] = -3.0e38f; l[r] = 0.f; }

    for (int kb0 = k0; kb0 < k1; kb0 += 32) {
      // ---- cooperative staging: K block (32x32 bf16) + V block (32x32 bf16)
      {
        if (tid < 128) {
          int row = tid >> 2, ch = tid & 3;
          const unsigned short* src =
              Kb + (bh * Nm + (size_t)(kb0 + row)) * HDm + ch * 8;
          __builtin_prefetch(src + 32 * HDm, 0, 0);
          STAGE16(src, sK + row * 40 + ch * 8);
        } else {
          int u = tid - 128;
          int row = u >> 2, ch = u & 3;
          const unsigned short* src =
              Vt + (bh * HDm + row) * Nm + kb0 + ch * 8;
          __builtin_prefetch(src + 32, 0, 0);
          STAGE16(src, sV + row * 40 + ch * 8);
        }
        wait_async0();
      }
      __syncthreads();

      if (active) {
        // ---- S = Q * K^T (two 16-key tiles) from LDS
        V16 kt1, kt2;
        {
          const unsigned short* kp1 = sK + lj * 40 + kbB;
          kt1.q[0] = *(const uint4*)kp1;
          kt1.q[1] = *(const uint4*)(kp1 + 16);
          const unsigned short* kp2 = sK + (lj + 16) * 40 + kbB;
          kt2.q[0] = *(const uint4*)kp2;
          kt2.q[1] = *(const uint4*)(kp2 + 16);
        }
        v8f z = {};
        v8f s1 = bfwmma(qa.v, kt1.v, z);
        v8f s2 = bfwmma(qa.v, kt2.v, z);

        if (kb0 + 32 > k1) {                     // remainder masking
          if (kb0 + lj >= k1) {
            #pragma unroll
            for (int r = 0; r < 8; ++r) s1[r] = -1.0e30f;
          }
          if (kb0 + 16 + lj >= k1) {
            #pragma unroll
            for (int r = 0; r < 8; ++r) s2[r] = -1.0e30f;
          }
        }

        // ---- online softmax (rows split lanes 0-15 / 16-31 per C layout)
        #pragma unroll
        for (int r = 0; r < 8; ++r) {
          float t  = rmax16(fmaxf(s1[r], s2[r]));
          float mn = fmaxf(m[r], t);
          float al = __expf(m[r] - mn);
          float p1 = __expf(s1[r] - mn);
          float p2 = __expf(s2[r] - mn);
          float srow = rsum16(p1 + p2);
          l[r] = l[r] * al + srow;
          acc1[r] *= al;
          acc2[r] *= al;
          m[r] = mn;
          int row = r + lhalf * 8;               // P -> LDS as bf16
          sPw[row * 40 + lj]      = f2bf_bits(p1);
          sPw[row * 40 + 16 + lj] = f2bf_bits(p2);
        }

        // ---- P back in A-operand layout (same-wave DS is in-order)
        V16 pa;
        {
          const unsigned short* pp = sPw + lj * 40 + kbA;
          pa.q[0] = *(const uint4*)pp;
          pa.q[1] = *(const uint4*)(pp + 16);
        }

        // ---- O += P * V from LDS (V transposed: row d, cols = keys)
        V16 vb1, vb2;
        {
          const unsigned short* vp1 = sV + lj * 40 + kbB;
          vb1.q[0] = *(const uint4*)vp1;
          vb1.q[1] = *(const uint4*)(vp1 + 16);
          const unsigned short* vp2 = sV + (lj + 16) * 40 + kbB;
          vb2.q[0] = *(const uint4*)vp2;
          vb2.q[1] = *(const uint4*)(vp2 + 16);
        }
        acc1 = bfwmma(pa.v, vb1.v, acc1);
        acc2 = bfwmma(pa.v, vb2.v, acc2);
      }
      __syncthreads();   // before restaging sK/sV
    }

    if (active) {
      #pragma unroll
      for (int r = 0; r < 8; ++r) {
        float inv = 1.0f / l[r];
        out1[r] += acc1[r] * inv;
        out2[r] += acc2[r] * inv;
      }
    }
  }

  if (active) {
    #pragma unroll
    for (int r = 0; r < 8; ++r) {
      int row = r + lhalf * 8;
      size_t t = (size_t)b * Nm + (size_t)(qt * 16 + row);
      O[t * Dm + h * HDm + lj]      = out1[r];
      O[t * Dm + h * HDm + 16 + lj] = out2[r];
    }
  }
}

// ---------------------------------------------------------------------------
extern "C" void kernel_launch(void* const* d_in, const int* in_sizes, int n_in,
                              void* d_out, int out_size, void* d_ws, size_t ws_size,
                              hipStream_t stream) {
  (void)in_sizes; (void)n_in; (void)out_size; (void)ws_size;
  const float* q    = (const float*)d_in[0];
  const float* k    = (const float*)d_in[1];
  const float* v    = (const float*)d_in[2];
  const float* fimg = (const float*)d_in[3];
  const float* ftxt = (const float*)d_in[4];
  const float* Wq   = (const float*)d_in[5];
  const float* bq   = (const float*)d_in[6];
  const float* Wk   = (const float*)d_in[7];
  const float* bk   = (const float*)d_in[8];
  const float* Wv   = (const float*)d_in[9];
  const float* bv   = (const float*)d_in[10];
  const float* Wo   = (const float*)d_in[11];
  const float* bo   = (const float*)d_in[12];
  // d_in[13] = num_k_exclude_rope (always 64 here; folded into NKEm)

  char*  ws  = (char*)d_ws;
  float* out = (float*)d_out;
  const unsigned short* Wb = (const unsigned short*)(ws + OFF_W);

  dim3 blk(256);
  wcvt_kernel<<<dim3(32, 4), blk, 0, stream>>>(Wq, Wk, Wv, Wo, ws);
  proj_kernel<<<dim3(MT), blk, 0, stream>>>(q, Wb + 0 * Dm * Dm, bq, fimg, ftxt,
                                            ws, nullptr, 0);
  proj_kernel<<<dim3(MT), blk, 0, stream>>>(k, Wb + 1 * Dm * Dm, bk, fimg, ftxt,
                                            ws, nullptr, 1);
  proj_kernel<<<dim3(MT), blk, 0, stream>>>(v, Wb + 2 * Dm * Dm, bv, fimg, ftxt,
                                            ws, nullptr, 2);
  attn_kernel<<<dim3((QT_PER_BH + 7) / 8, Bm * Hh), blk, 0, stream>>>(ws);
  proj_kernel<<<dim3(MT), blk, 0, stream>>>((const float*)(ws + OFF_O),
                                            Wb + 3 * Dm * Dm, bo, fimg, ftxt,
                                            ws, out, 3);
}
